// DKVMNSoftmax_58918361366955
// MI455X (gfx1250) — matrix operand order repaired and checked by reading, not compiled, not run on hardware
//
#include <hip/hip_runtime.h>
#include <hip/hip_bf16.h>

// ---------------------------------------------------------------------------
// DKVMN forward on gfx1250 (MI455X), wave32 + v_wmma_f32_16x16x32_f16.
//
// Dims: NQ=1000 K=5 M=50 KD=64 VD=64 SD=50 B=512 S=500, N = B*S = 256000.
// Kernel 1: per-wave 16-row tile: gather + attn-softmax + value_embed +
//           erase/add, all GEMMs via WMMA (f16 in, f32 acc).
// Kernel 2: per-batch-row scan, vm[50][64] in LDS, 500 steps.
// Kernel 3: summary GEMM (K=128) via WMMA + tanh + [50x5] classifier+softmax.
// Workspace: attn | erase | add | reads, each N*64 floats (stride 64).
// ---------------------------------------------------------------------------

#define NROWS  (512 * 500)     // 256000
#define SEQ    500
#define MMEM   50
#define KD     64
#define VD     64
#define SDIM   50
#define KCAT   5
#define STR    68              // LDS row stride (floats), conflict-avoiding

typedef _Float16 half16 __attribute__((ext_vector_type(16)));
typedef float    float8 __attribute__((ext_vector_type(8)));

union AFrag { half16 v; _Float16 h[16]; };
union CFrag { float8 v; float    f[8];  };

__device__ __forceinline__ float8 wmma_f16(half16 a, half16 b, float8 c) {
  // 8 args: (neg_a, A, neg_b, B, c_mod, C, reuse_a, reuse_b)
  return __builtin_amdgcn_wmma_f32_16x16x32_f16(
      false, a, false, b, (short)0, c, false, false);
}

// A fragment (16x32 f16) from LDS f32 tile buf[16][STR], k-base kb.
// Contiguous k order (same as B frags): lanes 0-15 -> M=lane, K=h;
//             lanes 16-31 -> M=lane-16, K = 16+h.
__device__ __forceinline__ half16 make_a_frag(const float (*buf)[STR],
                                              int lane, int kb) {
  AFrag a;
  const int m  = lane & 15;
  const int ko = (lane >> 4) << 4;
#pragma unroll
  for (int h = 0; h < 16; ++h) {
    const int k = h + ko;
    a.h[h] = (_Float16)buf[m][kb + k];
  }
  return a.v;
}

// B fragment (32x16 f16) from row-major weight W[kmax x ldn]: B(k,n)=W[k*ldn+n].
// Contiguous k order (same as A frag): lanes 0-15 -> K = h, lanes 16-31 -> K = 16+h; N = lane&15.
__device__ __forceinline__ half16 make_b_frag(const float* __restrict__ W,
                                              int ldn, int nmax, int kmax,
                                              int lane, int kb, int nb) {
  AFrag b;
  const int n  = nb + (lane & 15);
  const int ko = (lane >> 4) << 4;
#pragma unroll
  for (int h = 0; h < 16; ++h) {
    const int k = kb + ko + h;
    float v = (n < nmax && k < kmax) ? W[k * ldn + n] : 0.0f;
    b.h[h] = (_Float16)v;
  }
  return b.v;
}

// B fragment from transposed source: B(k,n) = Wt[n*ldk + k] (e.g. key_mem^T).
__device__ __forceinline__ half16 make_bT_frag(const float* __restrict__ Wt,
                                               int ldk, int nmax, int kmax,
                                               int lane, int kb, int nb) {
  AFrag b;
  const int n  = nb + (lane & 15);
  const int ko = (lane >> 4) << 4;
#pragma unroll
  for (int h = 0; h < 16; ++h) {
    const int k = kb + ko + h;
    float v = (n < nmax && k < kmax) ? Wt[n * ldk + k] : 0.0f;
    b.h[h] = (_Float16)v;
  }
  return b.v;
}

// ---------------------------------------------------------------------------
// Kernel 1: attn (softmax), value_embed, erase, add.  4 waves / block,
// each wave owns a 16-row tile.  grid = NROWS/64 blocks of 128 threads.
// ---------------------------------------------------------------------------
__global__ __launch_bounds__(128)
void k1_precompute(const int* __restrict__ questions,
                   const int* __restrict__ responses,
                   const float* __restrict__ qtab,
                   const float* __restrict__ itab,
                   const float* __restrict__ Wv, const float* __restrict__ bv,
                   const float* __restrict__ key_mem,
                   const float* __restrict__ We, const float* __restrict__ be,
                   const float* __restrict__ Wa, const float* __restrict__ ba,
                   float* __restrict__ attn_ws,
                   float* __restrict__ erase_ws,
                   float* __restrict__ add_ws) {
  __shared__ float bufA[4][16][STR];   // q_embed, later value_embed
  __shared__ float bufB[4][16][STR];   // item_v
  __shared__ float bufC[4][16][STR];   // attn logits staging
  __shared__ float respS[4][16][8];
  __shared__ int   qidxS[4][16];

  const int w    = threadIdx.x >> 5;
  const int lane = threadIdx.x & 31;
  const int tile = blockIdx.x * 4 + w;
  const int row0 = tile * 16;

  if (lane < 16) {
    const int r  = row0 + lane;
    qidxS[w][lane] = questions[r];
    const int rv = responses[r];
#pragma unroll
    for (int j = 0; j < KCAT; ++j) {
      float f = 1.0f - fabsf((float)j - (float)rv) * (1.0f / (KCAT - 1));
      respS[w][lane][j] = f > 0.0f ? f : 0.0f;
    }
  }
  __syncthreads();

  // Gather q_embed / item_v rows (16 x 64 each) into LDS.
#pragma unroll 4
  for (int j = 0; j < 32; ++j) {
    const int flat = lane + 32 * j;          // 0..1023
    const int i = flat >> 6, kd = flat & 63;
    const int q = qidxS[w][i];
    bufA[w][i][kd] = qtab[q * KD + kd];
    bufB[w][i][kd] = itab[q * KD + kd];
  }

  const half16 aq0 = make_a_frag(bufA[w], lane, 0);
  const half16 aq1 = make_a_frag(bufA[w], lane, 32);
  const half16 ai0 = make_a_frag(bufB[w], lane, 0);
  const half16 ai1 = make_a_frag(bufB[w], lane, 32);

  // ---- attn logits: q_embed @ key_mem^T  -> bufC[16][0..49] ----
#pragma unroll
  for (int nt = 0; nt < 4; ++nt) {
    float8 c = {};
    c = wmma_f16(aq0, make_bT_frag(key_mem, KD, MMEM, KD, lane, 0,  nt * 16), c);
    c = wmma_f16(aq1, make_bT_frag(key_mem, KD, MMEM, KD, lane, 32, nt * 16), c);
    CFrag cc; cc.v = c;
    const int n = nt * 16 + (lane & 15);
#pragma unroll
    for (int r = 0; r < 8; ++r) {
      const int m = r + ((lane >> 4) << 3);
      bufC[w][m][n] = cc.f[r];
    }
  }
  // softmax over M=50 per row, write attn to workspace
  if (lane < 16) {
    const int m = lane, row = row0 + m;
    float mx = -1e30f;
    for (int j = 0; j < MMEM; ++j) mx = fmaxf(mx, bufC[w][m][j]);
    float sum = 0.0f;
    for (int j = 0; j < MMEM; ++j) sum += __expf(bufC[w][m][j] - mx);
    const float inv = 1.0f / sum;
    for (int j = 0; j < MMEM; ++j)
      attn_ws[row * 64 + j] = __expf(bufC[w][m][j] - mx) * inv;
  }

  // ---- value_embed = [item_v | resp] @ Wv + bv  -> bufA (reused) ----
#pragma unroll
  for (int nt = 0; nt < 4; ++nt) {
    float8 c = {};
    c = wmma_f16(ai0, make_b_frag(Wv, VD, VD, KD, lane, 0,  nt * 16), c);
    c = wmma_f16(ai1, make_b_frag(Wv, VD, VD, KD, lane, 32, nt * 16), c);
    CFrag cc; cc.v = c;
    const int n = nt * 16 + (lane & 15);
    const float bn = bv[n];
#pragma unroll
    for (int r = 0; r < 8; ++r) {
      const int m = r + ((lane >> 4) << 3);
      float t = bn;
#pragma unroll
      for (int j = 0; j < KCAT; ++j)
        t += respS[w][m][j] * Wv[(KD + j) * VD + n];
      bufA[w][m][n] = cc.f[r] + t;
    }
  }

  const half16 av0 = make_a_frag(bufA[w], lane, 0);
  const half16 av1 = make_a_frag(bufA[w], lane, 32);

  // ---- erase = sigmoid(ve@We+be), add = tanh(ve@Wa+ba) ----
#pragma unroll
  for (int nt = 0; nt < 4; ++nt) {
    const int n = nt * 16 + (lane & 15);
    float8 ce = {};
    ce = wmma_f16(av0, make_b_frag(We, VD, VD, KD, lane, 0,  nt * 16), ce);
    ce = wmma_f16(av1, make_b_frag(We, VD, VD, KD, lane, 32, nt * 16), ce);
    float8 ca = {};
    ca = wmma_f16(av0, make_b_frag(Wa, VD, VD, KD, lane, 0,  nt * 16), ca);
    ca = wmma_f16(av1, make_b_frag(Wa, VD, VD, KD, lane, 32, nt * 16), ca);
    CFrag ue, ua; ue.v = ce; ua.v = ca;
    const float ben = be[n], ban = ba[n];
#pragma unroll
    for (int r = 0; r < 8; ++r) {
      const int m = r + ((lane >> 4) << 3);
      const int row = row0 + m;
      erase_ws[row * 64 + n] = 1.0f / (1.0f + __expf(-(ue.f[r] + ben)));
      add_ws[row * 64 + n]   = tanhf(ua.f[r] + ban);
    }
  }
}

// ---------------------------------------------------------------------------
// Kernel 2: sequential memory scan. One block per batch row b, 64 threads;
// thread v owns column v of vm[50][64] (conflict-free LDS).
// ---------------------------------------------------------------------------
__global__ __launch_bounds__(64)
void k2_scan(const float* __restrict__ attn_ws,
             const float* __restrict__ erase_ws,
             const float* __restrict__ add_ws,
             const float* __restrict__ init_vm,
             float* __restrict__ reads_ws) {
  __shared__ float vm[MMEM][64];
  __shared__ float a_sh[MMEM + 2];
  const int v = threadIdx.x;
  const int b = blockIdx.x;

  for (int m = 0; m < MMEM; ++m) vm[m][v] = init_vm[m * VD + v];
  __syncthreads();

  for (int t = 0; t < SEQ; ++t) {
    const int n = b * SEQ + t;
    if (v < MMEM) a_sh[v] = attn_ws[n * 64 + v];
    __syncthreads();
    const float e  = erase_ws[n * 64 + v];
    const float ad = add_ws[n * 64 + v];
    float rd = 0.0f;
#pragma unroll 10
    for (int m = 0; m < MMEM; ++m) {
      const float a = a_sh[m];
      const float o = vm[m][v];
      rd += a * o;
      vm[m][v] = o * (1.0f - a * e) + a * ad;   // read uses old vm; column-private
    }
    reads_ws[n * 64 + v] = rd;
    __syncthreads();
  }
}

// ---------------------------------------------------------------------------
// Kernel 3: summary = tanh([reads|q_embed] @ Ws + bs) via WMMA (K=128),
// then logits = summary @ Wc + bc, softmax -> probs. 4 waves/block.
// ---------------------------------------------------------------------------
__global__ __launch_bounds__(128)
void k3_summary(const int* __restrict__ questions,
                const float* __restrict__ qtab,
                const float* __restrict__ reads_ws,
                const float* __restrict__ Ws, const float* __restrict__ bs,
                const float* __restrict__ Wc, const float* __restrict__ bc,
                float* __restrict__ out) {
  __shared__ float bufA[4][16][STR];   // reads
  __shared__ float bufB[4][16][STR];   // q_embed
  __shared__ float bufC[4][16][STR];   // summary
  const int w    = threadIdx.x >> 5;
  const int lane = threadIdx.x & 31;
  const int tile = blockIdx.x * 4 + w;
  const int row0 = tile * 16;

#pragma unroll 4
  for (int j = 0; j < 32; ++j) {
    const int flat = lane + 32 * j;
    const int i = flat >> 6, kd = flat & 63;
    const int r = row0 + i;
    bufA[w][i][kd] = reads_ws[r * 64 + kd];
    const int q = questions[r];
    bufB[w][i][kd] = qtab[q * KD + kd];
  }

  const half16 a0 = make_a_frag(bufA[w], lane, 0);
  const half16 a1 = make_a_frag(bufA[w], lane, 32);
  const half16 a2 = make_a_frag(bufB[w], lane, 0);
  const half16 a3 = make_a_frag(bufB[w], lane, 32);

#pragma unroll
  for (int nt = 0; nt < 4; ++nt) {
    float8 c = {};
    c = wmma_f16(a0, make_b_frag(Ws, SDIM, SDIM, 128, lane, 0,  nt * 16), c);
    c = wmma_f16(a1, make_b_frag(Ws, SDIM, SDIM, 128, lane, 32, nt * 16), c);
    c = wmma_f16(a2, make_b_frag(Ws, SDIM, SDIM, 128, lane, 64, nt * 16), c);
    c = wmma_f16(a3, make_b_frag(Ws, SDIM, SDIM, 128, lane, 96, nt * 16), c);
    CFrag cc; cc.v = c;
    const int n = nt * 16 + (lane & 15);
#pragma unroll
    for (int r = 0; r < 8; ++r) {
      const int m = r + ((lane >> 4) << 3);
      bufC[w][m][n] = (n < SDIM) ? tanhf(cc.f[r] + bs[n]) : 0.0f;
    }
  }

  if (lane < 16) {
    const int m = lane, row = row0 + m;
    float lg[KCAT];
#pragma unroll
    for (int j = 0; j < KCAT; ++j) lg[j] = bc[j];
    for (int i = 0; i < SDIM; ++i) {
      const float s = bufC[w][m][i];
#pragma unroll
      for (int j = 0; j < KCAT; ++j) lg[j] += s * Wc[i * KCAT + j];
    }
    float mx = lg[0];
#pragma unroll
    for (int j = 1; j < KCAT; ++j) mx = fmaxf(mx, lg[j]);
    float ex[KCAT], sum = 0.0f;
#pragma unroll
    for (int j = 0; j < KCAT; ++j) { ex[j] = __expf(lg[j] - mx); sum += ex[j]; }
    const float inv = 1.0f / sum;
#pragma unroll
    for (int j = 0; j < KCAT; ++j) {
      out[row * KCAT + j]                     = lg[j];        // logits
      out[NROWS * KCAT + row * KCAT + j]      = ex[j] * inv;  // probs
    }
  }
}

// ---------------------------------------------------------------------------
extern "C" void kernel_launch(void* const* d_in, const int* in_sizes, int n_in,
                              void* d_out, int out_size, void* d_ws,
                              size_t ws_size, hipStream_t stream) {
  (void)in_sizes; (void)n_in; (void)out_size; (void)ws_size;

  const int*   questions = (const int*)  d_in[0];
  const int*   responses = (const int*)  d_in[1];
  const float* qtab      = (const float*)d_in[2];
  const float* itab      = (const float*)d_in[3];
  const float* Wv        = (const float*)d_in[4];
  const float* bv        = (const float*)d_in[5];
  const float* key_mem   = (const float*)d_in[6];
  const float* init_vm   = (const float*)d_in[7];
  const float* We        = (const float*)d_in[8];
  const float* be        = (const float*)d_in[9];
  const float* Wa        = (const float*)d_in[10];
  const float* ba        = (const float*)d_in[11];
  const float* Ws        = (const float*)d_in[12];
  const float* bs        = (const float*)d_in[13];
  const float* Wc        = (const float*)d_in[14];
  const float* bc        = (const float*)d_in[15];

  float* attn_ws  = (float*)d_ws;                       // N*64
  float* erase_ws = attn_ws  + (size_t)NROWS * 64;      // N*64
  float* add_ws   = erase_ws + (size_t)NROWS * 64;      // N*64
  float* reads_ws = add_ws   + (size_t)NROWS * 64;      // N*64
  float* out      = (float*)d_out;

  const int tiles  = NROWS / 16;      // 16000
  const int blocks = tiles / 4;       // 4000 (4 waves of one 16-row tile each)

  k1_precompute<<<blocks, 128, 0, stream>>>(
      questions, responses, qtab, itab, Wv, bv, key_mem,
      We, be, Wa, ba, attn_ws, erase_ws, add_ws);

  k2_scan<<<512, 64, 0, stream>>>(attn_ws, erase_ws, add_ws, init_vm, reads_ws);

  k3_summary<<<blocks, 128, 0, stream>>>(
      questions, qtab, reads_ws, Ws, bs, Wc, bc, out);
}
